// MambaTTSDecoder_61478161875536
// MI455X (gfx1250) — matrix operand order repaired and checked by reading, not compiled, not run on hardware
//
#include <hip/hip_runtime.h>
#include <hip/hip_bf16.h>
#include <math.h>

// ---------------------------------------------------------------------------
// MambaTTSDecoder forward for MI455X (gfx1250).
// GEMMs run on v_wmma_f32_16x16x32_bf16 with pre-converted bf16 operands so
// the inner loop is pure global_load_b128 + v_wmma (no per-tile cvt VALU).
// ---------------------------------------------------------------------------

#define Bsz   4
#define Tlen  2048
#define Dm    512
#define Dff   2048
#define Di    1024      // d_inner
#define Ds    16        // d_state
#define Dr    32        // dt_rank
#define TFs   512       // styled frames
#define NH    8
#define HD    64
#define VOC   1024

typedef __attribute__((ext_vector_type(16))) __bf16 v16bf;
typedef __attribute__((ext_vector_type(8)))  __bf16 v8bf;
typedef __attribute__((ext_vector_type(8)))  float  v8f;

__device__ __forceinline__ __bf16 f2bf(float f) {
  union { float f; unsigned u; } a; a.f = f;
  unsigned r = a.u + 0x7fffu + ((a.u >> 16) & 1u);   // round-to-nearest-even
  union { unsigned short s; __bf16 b; } c;
  c.s = (unsigned short)(r >> 16);
  return c.b;
}

// ---------------------------------------------------------------------------
// f32 -> bf16 bulk conversion (4 elems/thread). n4 = n/4.
// ---------------------------------------------------------------------------
__global__ __launch_bounds__(256)
void cvt_kernel(const float* __restrict__ in, __bf16* __restrict__ out, int n4)
{
  int i = blockIdx.x * blockDim.x + threadIdx.x;
  if (i >= n4) return;
  float4 f = ((const float4*)in)[i];
  union { __bf16 b[4]; unsigned long long u; } o;
  o.b[0] = f2bf(f.x); o.b[1] = f2bf(f.y); o.b[2] = f2bf(f.z); o.b[3] = f2bf(f.w);
  ((unsigned long long*)out)[i] = o.u;
}

// ---------------------------------------------------------------------------
// WMMA GEMM:  out[M,N] = act(A[M,K] @ W[N,K]^T + bias) + res
//   A: bf16, row stride lda.  W: bf16 (N,K).  out (f32) / outb (bf16) optional.
//   M%32==0, N%64==0, K%32==0.  Wave tile = 32x64 (2x4 WMMA tiles).
// CDNA5 bf16 lane layouts:
//   A(16x32): lane = row(M)%16 [+16 -> K+8 half], halves {hi*8+0..7, 16+hi*8+0..7}
//   B(32x16): lane = col(N)%16 [+16 -> K+8 half], halves {hi*8+0..7, 16+hi*8+0..7}
//   C/D 16x16: lane L, vgpr p -> row = (L>=16?8:0)+p, col = L%16
// act: 0=none, 1=softplus, 2=exact gelu
// ---------------------------------------------------------------------------
__global__ __launch_bounds__(256)
void gemm_wmma_kernel(const __bf16* __restrict__ A, int lda,
                      const __bf16* __restrict__ W,
                      const float* __restrict__ bias,
                      const float* __restrict__ res,
                      float* __restrict__ out,
                      __bf16* __restrict__ outb,
                      int M, int N, int K, int act)
{
  const int lane = threadIdx.x & 31;
  const int wave = threadIdx.x >> 5;
  const int tilesN = N >> 6;
  int tile = blockIdx.x * (blockDim.x >> 5) + wave;
  if (tile >= (M >> 5) * tilesN) return;          // wave-uniform: EXEC all-1 below
  const int m0 = (tile / tilesN) << 5;
  const int n0 = (tile % tilesN) << 6;
  const int r  = lane & 15;
  const int hi = lane >> 4;

  v8f acc[2][4] = {};
  const __bf16* ap0 = A + (size_t)(m0 + r) * lda + hi * 8;
  const __bf16* ap1 = ap0 + (size_t)16 * lda;
  const __bf16* bp  = W + (size_t)(n0 + r) * K + hi * 8;

  for (int k0 = 0; k0 < K; k0 += 32) {
    v16bf a[2], b[4];
#pragma unroll
    for (int i = 0; i < 2; ++i) {
      const __bf16* p = (i ? ap1 : ap0) + k0;
      v8bf lo = *(const v8bf*)p;
      v8bf hh = *(const v8bf*)(p + 16);
      a[i] = __builtin_shufflevector(lo, hh, 0,1,2,3,4,5,6,7,8,9,10,11,12,13,14,15);
    }
#pragma unroll
    for (int j = 0; j < 4; ++j) {
      const __bf16* p = bp + (size_t)j * 16 * K + k0;
      v8bf lo = *(const v8bf*)p;
      v8bf hh = *(const v8bf*)(p + 16);
      b[j] = __builtin_shufflevector(lo, hh, 0,1,2,3,4,5,6,7,8,9,10,11,12,13,14,15);
    }
#pragma unroll
    for (int i = 0; i < 2; ++i)
#pragma unroll
      for (int j = 0; j < 4; ++j)
        acc[i][j] = __builtin_amdgcn_wmma_f32_16x16x32_bf16(
            false, a[i], false, b[j], (short)0, acc[i][j], false, false);
  }

#pragma unroll
  for (int i = 0; i < 2; ++i) {
#pragma unroll
    for (int j = 0; j < 4; ++j) {
      int col = n0 + j * 16 + r;
      float bv = bias ? bias[col] : 0.f;
#pragma unroll
      for (int p = 0; p < 8; ++p) {
        int row = m0 + i * 16 + hi * 8 + p;
        float v = acc[i][j][p] + bv;
        if (act == 1) v = (v > 20.f) ? v : log1pf(__expf(v));                 // softplus
        else if (act == 2) v = 0.5f * v * (1.f + erff(v * 0.70710678118f));   // exact gelu
        size_t o = (size_t)row * N + col;
        if (res) v += res[o];
        if (out)  out[o]  = v;
        if (outb) outb[o] = f2bf(v);
      }
    }
  }
}

// ---------------------------------------------------------------------------
// x = token_embed[tok] + pos_embed[t] + quant_embed[0]
// ---------------------------------------------------------------------------
__global__ __launch_bounds__(256)
void embed_kernel(const int* __restrict__ tok,
                  const float* __restrict__ te,
                  const float* __restrict__ pe,
                  const float* __restrict__ qe,
                  float* __restrict__ x)
{
  size_t i = (size_t)blockIdx.x * blockDim.x + threadIdx.x;  // B*T*Dm
  int d = (int)(i & (Dm - 1));
  size_t bt = i >> 9;
  int t = (int)(bt & (Tlen - 1));
  int id = tok[bt];
  x[i] = te[(size_t)id * Dm + d] + pe[(size_t)t * Dm + d] + qe[d];
}

// ---------------------------------------------------------------------------
// LayerNorm over last dim (512); writes bf16 (consumed only as GEMM A).
// ---------------------------------------------------------------------------
__global__ __launch_bounds__(256)
void layernorm_kernel(const float* __restrict__ x,
                      const float* __restrict__ w,
                      const float* __restrict__ bvec,
                      __bf16* __restrict__ out)
{
  __shared__ float s1[256], s2[256];
  size_t row = blockIdx.x;
  int tid = threadIdx.x;
  const float* xr = x + row * Dm;
  float a = xr[tid], b = xr[tid + 256];
  s1[tid] = a + b;
  s2[tid] = a * a + b * b;
  __syncthreads();
  for (int st = 128; st > 0; st >>= 1) {
    if (tid < st) { s1[tid] += s1[tid + st]; s2[tid] += s2[tid + st]; }
    __syncthreads();
  }
  float mu  = s1[0] * (1.f / Dm);
  float var = s2[0] * (1.f / Dm) - mu * mu;
  float rs  = rsqrtf(var + 1e-5f);
  out[row * Dm + tid]       = f2bf((a - mu) * rs * w[tid]       + bvec[tid]);
  out[row * Dm + tid + 256] = f2bf((b - mu) * rs * w[tid + 256] + bvec[tid + 256]);
}

// ---------------------------------------------------------------------------
// Causal depthwise conv (K=4) + bias + SiLU. Writes f32 (scan) and bf16 (GEMM A).
// ---------------------------------------------------------------------------
__global__ __launch_bounds__(256)
void conv_silu_kernel(const float* __restrict__ xz,
                      const float* __restrict__ cw,
                      const float* __restrict__ cb,
                      float* __restrict__ xc,
                      __bf16* __restrict__ xcb)
{
  size_t i = (size_t)blockIdx.x * blockDim.x + threadIdx.x;  // B*T*Di
  int c = (int)(i & (Di - 1));
  size_t bt = i >> 10;
  int t = (int)(bt & (Tlen - 1));
  size_t brow = bt - t;  // b*T
  float acc = cb[c];
#pragma unroll
  for (int j = 0; j < 4; ++j) {
    int tt = t - 3 + j;
    if (tt >= 0) acc += xz[(brow + tt) * (size_t)(2 * Di) + c] * cw[c * 4 + j];
  }
  float s = acc / (1.f + __expf(-acc));   // silu
  xc[i] = s;
  xcb[i] = f2bf(s);
}

// ---------------------------------------------------------------------------
// Selective scan: one thread per (b, channel); 16 states in registers.
// Fuses D-skip + silu(z) gating; writes gated y as bf16 (GEMM A for out_proj).
// ---------------------------------------------------------------------------
__global__ __launch_bounds__(256)
void scan_kernel(const float* __restrict__ xc,
                 const float* __restrict__ dt,
                 const float* __restrict__ xdbl,
                 const float* __restrict__ xz,
                 const float* __restrict__ A_log,
                 const float* __restrict__ Dsk,
                 __bf16* __restrict__ yb)
{
  int idx = blockIdx.x * blockDim.x + threadIdx.x;  // B*Di threads
  int c = idx & (Di - 1);
  int b = idx >> 10;
  float A[Ds], s[Ds];
#pragma unroll
  for (int n = 0; n < Ds; ++n) { A[n] = -__expf(A_log[c * Ds + n]); s[n] = 0.f; }
  float Dc = Dsk[c];
  for (int t = 0; t < Tlen; ++t) {
    size_t bt = (size_t)b * Tlen + t;
    float xt = xc[bt * Di + c];
    float dv = dt[bt * Di + c];
    const float* Bc = xdbl + bt * 64 + Dr;   // x_dbl[:, 32:48]
    const float* Cc = Bc + Ds;               // x_dbl[:, 48:64]
    float dx = dv * xt;
    float y = 0.f;
#pragma unroll
    for (int n = 0; n < Ds; ++n) {
      s[n] = __expf(dv * A[n]) * s[n] + dx * Bc[n];
      y += s[n] * Cc[n];
    }
    float zt = xz[bt * (size_t)(2 * Di) + Di + c];
    float g = zt / (1.f + __expf(-zt));
    yb[bt * Di + c] = f2bf((y + Dc * xt) * g);
  }
}

// ---------------------------------------------------------------------------
// Fused cross-attention, online softmax. One wave per query; writes bf16 ctx.
// ---------------------------------------------------------------------------
__global__ __launch_bounds__(256)
void attn_kernel(const float* __restrict__ q,
                 const float* __restrict__ k,
                 const float* __restrict__ v,
                 const unsigned char* __restrict__ mask,
                 __bf16* __restrict__ outb)
{
  int lane = threadIdx.x & 31;
  int gw = blockIdx.x * 8 + (threadIdx.x >> 5);   // one query per wave
  int t  = gw % Tlen;
  int bh = gw / Tlen;
  int b  = bh / NH;
  int h  = bh % NH;
  size_t qoff = ((size_t)(b * Tlen + t)) * Dm + h * HD;
  float q0 = q[qoff + lane];
  float q1 = q[qoff + lane + 32];
  float m = -3.4e38f, l = 0.f, o0 = 0.f, o1 = 0.f;
  for (int s = 0; s < TFs; ++s) {
    size_t koff = ((size_t)(b * TFs + s)) * Dm + h * HD;
    float p = q0 * k[koff + lane] + q1 * k[koff + lane + 32];
#pragma unroll
    for (int off = 16; off > 0; off >>= 1) p += __shfl_xor(p, off);
    float sc = p * 0.125f;                        // 1/sqrt(64)
    if (!mask[b * TFs + s]) sc = -1e9f;
    float mn = fmaxf(m, sc);
    float ex = __expf(sc - mn);
    float cf = __expf(m - mn);
    l  = l  * cf + ex;
    o0 = o0 * cf + ex * v[koff + lane];
    o1 = o1 * cf + ex * v[koff + lane + 32];
    m = mn;
  }
  outb[qoff + lane]      = f2bf(o0 / l);
  outb[qoff + lane + 32] = f2bf(o1 / l);
}

// ---------------------------------------------------------------------------
static void gemm(hipStream_t s, const __bf16* A, int lda,
                 const float* Wf, __bf16* Wb,
                 const float* bias, const float* res,
                 float* out, __bf16* outb,
                 int M, int N, int K, int act)
{
  int n4 = (N * K) / 4;
  cvt_kernel<<<(n4 + 255) / 256, 256, 0, s>>>(Wf, Wb, n4);
  int tiles = (M / 32) * (N / 64);
  gemm_wmma_kernel<<<(tiles + 7) / 8, 256, 0, s>>>(A, lda, Wb, bias, res, out, outb, M, N, K, act);
}

// Param flattening: setup_inputs() recursive insertion order.
// 0 audio_tokens, 1 styled_frames, 2 styled_mask,
// 3 token_embed, 4 pos_embed, 5 quant_embed, 6 norm_out_w, 7 norm_out_b,
// 8 head_w, 9 head_b, then 6 layers x 27 tensors each.
extern "C" void kernel_launch(void* const* d_in, const int* in_sizes, int n_in,
                              void* d_out, int out_size, void* d_ws, size_t ws_size,
                              hipStream_t stream)
{
  const int*   tokens = (const int*)d_in[0];
  const float* frames = (const float*)d_in[1];
  const unsigned char* mask = (const unsigned char*)d_in[2];
  const float* te  = (const float*)d_in[3];
  const float* pe  = (const float*)d_in[4];
  const float* qe  = (const float*)d_in[5];
  const float* now = (const float*)d_in[6];
  const float* nob = (const float*)d_in[7];
  const float* hw  = (const float*)d_in[8];
  const float* hb  = (const float*)d_in[9];

  // ---- workspace: f32 region then bf16 region (~191 MB total) ----
  float* x    = (float*)d_ws;                     // (8192, 512)
  float* xz   = x    + (size_t)4194304;           // (8192, 2048)
  float* xc   = xz   + (size_t)16777216;          // (8192, 1024); qb reuses
  float* xdbl = xc   + (size_t)8388608;           // (8192, 64)
  float* dtb  = xdbl + (size_t)524288;            // (8192, 1024); kb|vb reuse
  __bf16* hbuf  = (__bf16*)(dtb + (size_t)8388608);
  __bf16* R1    = hbuf  + (size_t)4194304;        // 16M halves
  __bf16* xdblb = R1    + (size_t)16777216;
  __bf16* fb    = xdblb + (size_t)524288;
  __bf16* Wb    = fb    + (size_t)1048576;
  __bf16* xcb   = R1;                             // (8192,1024) bf16
  __bf16* yb    = R1 + (size_t)8388608;           // (8192,1024) bf16
  __bf16* ffb   = R1;                             // (8192,2048) bf16 (after xcb/yb dead)
  __bf16* ctxb  = hbuf;                           // (8192,512) bf16 (after wq consumed h)
  float*  qb    = xc;                             // f32 reuse after scan
  float*  kb    = dtb;
  float*  vb    = dtb + (size_t)1048576;

  const int Mq = Bsz * Tlen;   // 8192
  const int Mk = Bsz * TFs;    // 2048

  embed_kernel<<<16384, 256, 0, stream>>>(tokens, te, pe, qe, x);
  cvt_kernel<<<(Mk * Dm / 4 + 255) / 256, 256, 0, stream>>>(frames, fb, Mk * Dm / 4);

  for (int l = 0; l < 6; ++l) {
    const float* const* P = (const float* const*)(d_in + 10 + l * 27);
    const float *ln_mw = P[0], *ln_mb = P[1], *ln_cw = P[2], *ln_cb = P[3];
    const float *ln_fw = P[4], *ln_fb = P[5];
    const float *inw = P[6], *cw = P[7], *cb = P[8], *xpw = P[9];
    const float *dtw = P[10], *dtpb = P[11], *alog = P[12], *dsk = P[13], *opw = P[14];
    const float *wq = P[15], *wqb = P[16], *wk = P[17], *wkb = P[18];
    const float *wv = P[19], *wvb = P[20], *wo = P[21], *wob = P[22];
    const float *f1w = P[23], *f1b = P[24], *f2w = P[25], *f2b = P[26];

    // ---- Mamba block ----
    layernorm_kernel<<<Mq, 256, 0, stream>>>(x, ln_mw, ln_mb, hbuf);
    gemm(stream, hbuf, Dm, inw, Wb, nullptr, nullptr, xz, nullptr, Mq, 2 * Di, Dm, 0);
    conv_silu_kernel<<<32768, 256, 0, stream>>>(xz, cw, cb, xc, xcb);
    gemm(stream, xcb, Di, xpw, Wb, nullptr, nullptr, xdbl, xdblb, Mq, 64, Di, 0);
    gemm(stream, xdblb, 64, dtw, Wb, dtpb, nullptr, dtb, nullptr, Mq, Di, Dr, 1); // softplus
    scan_kernel<<<16, 256, 0, stream>>>(xc, dtb, xdbl, xz, alog, dsk, yb);
    gemm(stream, yb, Di, opw, Wb, nullptr, x, x, nullptr, Mq, Dm, Di, 0);         // + residual

    // ---- Cross attention ----
    layernorm_kernel<<<Mq, 256, 0, stream>>>(x, ln_cw, ln_cb, hbuf);
    gemm(stream, hbuf, Dm, wq, Wb, wqb, nullptr, qb, nullptr, Mq, Dm, Dm, 0);
    gemm(stream, fb, Dm, wk, Wb, wkb, nullptr, kb, nullptr, Mk, Dm, Dm, 0);
    gemm(stream, fb, Dm, wv, Wb, wvb, nullptr, vb, nullptr, Mk, Dm, Dm, 0);
    attn_kernel<<<8192, 256, 0, stream>>>(qb, kb, vb, mask, ctxb);
    gemm(stream, ctxb, Dm, wo, Wb, wob, x, x, nullptr, Mq, Dm, Dm, 0);            // + residual

    // ---- FFN ----
    layernorm_kernel<<<Mq, 256, 0, stream>>>(x, ln_fw, ln_fb, hbuf);
    gemm(stream, hbuf, Dm, f1w, Wb, f1b, nullptr, nullptr, ffb, Mq, Dff, Dm, 2);  // gelu
    gemm(stream, ffb, Dff, f2w, Wb, f2b, x, x, nullptr, Mq, Dm, Dff, 0);          // + residual
  }

  layernorm_kernel<<<Mq, 256, 0, stream>>>(x, now, nob, hbuf);
  gemm(stream, hbuf, Dm, hw, Wb, hb, nullptr, (float*)d_out, nullptr, Mq, VOC, Dm, 0);
}
